// GNNLSTMModel_523986010199
// MI455X (gfx1250) — matrix operand order repaired and checked by reading, not compiled, not run on hardware
//
#include <hip/hip_runtime.h>

typedef __attribute__((ext_vector_type(16))) _Float16 v16h;
typedef __attribute__((ext_vector_type(8)))  float    v8f;

// ---------------- structural constants (match reference) ----------------
constexpr int Fdim = 128;   // input features
constexpr int Gdim = 32;    // gnn hidden
constexpr int Hd   = 50;    // lstm hidden
constexpr int H4   = 200;   // 4*H
constexpr int NT   = 13;    // 4H padded to 208 = NT*16 column tiles
constexpr int NCOL = 208;
constexpr int KT_H = 2;     // H=50 padded to 64 = KT_H*32

__device__ __forceinline__ v8f wmma16(v16h a, v16h b, v8f c) {
  return __builtin_amdgcn_wmma_f32_16x16x32_f16(
      /*neg_a=*/false, a, /*neg_b=*/false, b,
      /*c_mod=*/(short)0, c, /*reuse_a=*/false, /*reuse_b=*/false);
}

// Load this lane's B fragment for a given tile. Fragment layout (from pack_b):
//   frag[(tile*32 + lane)*16 + e]  -> per-lane 32B contiguous, 32B aligned.
__device__ __forceinline__ v16h load_bfrag(const _Float16* frag, int tile) {
  const int lane = threadIdx.x & 31;
  return *(const v16h*)(frag + (size_t)tile * 512 + lane * 16);
}

// Build an A fragment (16x32 f16 layout, ISA 7.12.2) from f32 LDS rows.
__device__ __forceinline__ v16h make_afrag_lds(const float* S, int ldw, int kt) {
  const int lane = threadIdx.x & 31;
  const int m  = lane & 15;
  const int kb = kt * 32 + ((lane & 16) ? 8 : 0);
  v16h a;
#pragma unroll
  for (int e = 0; e < 8; ++e) a[e]     = (_Float16)S[m * ldw + kb + e];
#pragma unroll
  for (int e = 0; e < 8; ++e) a[8 + e] = (_Float16)S[m * ldw + kb + 16 + e];
  return a;
}

__device__ __forceinline__ float sigmoidf(float v) {
  return 1.0f / (1.0f + __expf(-v));
}

// ---------------- weight fragment packing ----------------
// B fragment layout (16-bit, 32x16): lane holds column n = lane&15;
// lanes 0-15 hold K=0..7,16..23, lanes 16-31 K=8..15,24..31; slot e -> K = base + e (+8 if e>=8).
// frag[((kt*NTt + nt)*32 + lane)*16 + e]
__global__ void pack_b(const float* __restrict__ W, _Float16* __restrict__ frag,
                       int K, int Ncols, int ld, int trans, int KT, int NTt) {
  int idx = blockIdx.x * blockDim.x + threadIdx.x;
  int total = KT * NTt * 512;
  if (idx >= total) return;
  int e    = idx & 15;
  int lane = (idx >> 4) & 31;
  int tile = idx >> 9;
  int nt   = tile % NTt;
  int kt   = tile / NTt;
  int k = kt * 32 + ((lane & 16) ? 8 : 0) + e + ((e >= 8) ? 8 : 0);
  int n = nt * 16 + (lane & 15);
  float v = 0.0f;
  if (k < K && n < Ncols)
    v = trans ? W[(size_t)n * ld + k] : W[(size_t)k * ld + n];
  frag[idx] = (_Float16)v;
}

// ---------------- degree / normalization ----------------
__global__ void init_deg(float* deg, int N) {
  int i = blockIdx.x * blockDim.x + threadIdx.x;
  if (i < N) deg[i] = 1.0f;  // self-loop
}
__global__ void count_deg(const long long* __restrict__ ei, float* deg, int E) {
  int i = blockIdx.x * blockDim.x + threadIdx.x;
  if (i < E) atomicAdd(&deg[(int)ei[(size_t)E + i]], 1.0f);
}
__global__ void finalize_dinv(float* deg, int N) {
  int i = blockIdx.x * blockDim.x + threadIdx.x;
  if (i < N) deg[i] = rsqrtf(deg[i]);  // deg >= 1 always (self loop)
}

// ---------------- GCN linear: h = x @ W_gcn  (WMMA f16) ----------------
__global__ __launch_bounds__(128) void gcn_mm(const float* __restrict__ x,
                                              const _Float16* __restrict__ fw,
                                              float* __restrict__ h, int N) {
  const int wave = threadIdx.x >> 5;
  const int lane = threadIdx.x & 31;
  const int rowBase = blockIdx.x * 64 + wave * 16;
  if (rowBase >= N) return;  // wave-uniform
  const int m   = lane & 15;
  const int kb0 = (lane & 16) ? 8 : 0;
  int row = rowBase + m;
  if (row >= N) row = N - 1;  // clamp read; store guarded below
  const float* xr = x + (size_t)row * Fdim;
  v8f c0 = {}, c1 = {};
#pragma unroll
  for (int kt = 0; kt < 4; ++kt) {
    const int kb = kt * 32 + kb0;
    v16h a;
#pragma unroll
    for (int e = 0; e < 8; ++e) a[e]     = (_Float16)xr[kb + e];
#pragma unroll
    for (int e = 0; e < 8; ++e) a[8 + e] = (_Float16)xr[kb + 16 + e];
    c0 = wmma16(a, load_bfrag(fw, kt * 2 + 0), c0);
    c1 = wmma16(a, load_bfrag(fw, kt * 2 + 1), c1);
  }
  const int n   = lane & 15;
  const int mof = (lane & 16) ? 8 : 0;
#pragma unroll
  for (int r = 0; r < 8; ++r) {
    int rr = rowBase + mof + r;
    if (rr < N) {
      h[(size_t)rr * Gdim + n]      = c0[r];
      h[(size_t)rr * Gdim + 16 + n] = c1[r];
    }
  }
}

// ---------------- self-loop init + bias ----------------
__global__ void self_loop_bias(const float* __restrict__ h,
                               const float* __restrict__ dinv,
                               const float* __restrict__ b,
                               float* __restrict__ gnn, int N) {
  int idx = blockIdx.x * blockDim.x + threadIdx.x;
  if (idx < N * Gdim) {
    int i = idx >> 5, f = idx & 31;
    float di = dinv[i];
    gnn[idx] = h[idx] * di * di + b[f];
  }
}

// ---------------- edge gather/scale/scatter (wave per edge) ----------------
__global__ void edge_scatter(const long long* __restrict__ ei,
                             const float* __restrict__ h,
                             const float* __restrict__ dinv,
                             float* __restrict__ gnn, int E) {
  int lane = threadIdx.x & 31;
  int e = (blockIdx.x * blockDim.x + threadIdx.x) >> 5;
  if (e >= E) return;
  int s = (int)ei[e];
  int d = (int)ei[(size_t)E + e];
  float norm = dinv[s] * dinv[d];
  atomicAdd(&gnn[(size_t)d * Gdim + lane], h[(size_t)s * Gdim + lane] * norm);
}

// ---------------- fused 2-layer LSTM + FC (16 nodes / block) ----------------
__global__ __launch_bounds__(128) void lstm_mega(
    const float* __restrict__ gnn,
    const _Float16* __restrict__ fih0, const _Float16* __restrict__ fhh0,
    const _Float16* __restrict__ fih1, const _Float16* __restrict__ fhh1,
    const float* __restrict__ bih0, const float* __restrict__ bhh0,
    const float* __restrict__ bih1, const float* __restrict__ bhh1,
    const float* __restrict__ fcw, const float* __restrict__ fcb,
    const int* __restrict__ seqp, float* __restrict__ out, int N) {
  __shared__ float sH0[16 * 64], sC0[16 * 64], sH1[16 * 64], sC1[16 * 64];
  __shared__ float sP0[16 * NCOL], sG[16 * NCOL];
  const int tid  = threadIdx.x;
  const int wave = tid >> 5;
  const int lane = tid & 31;
  const int nodeBase = blockIdx.x * 16;

  for (int j = tid; j < 16 * 64; j += 128) {
    sH0[j] = 0.f; sC0[j] = 0.f; sH1[j] = 0.f; sC1[j] = 0.f;
  }
  __syncthreads();

  // p0 = gnn @ w_ih0.T + (b_ih0 + b_hh0)   (K = 32, single k-tile)
  {
    const int m  = lane & 15;
    const int kb = (lane & 16) ? 8 : 0;
    int row = nodeBase + m;
    if (row >= N) row = N - 1;
    const float* gr = gnn + (size_t)row * Gdim;
    v16h a;
#pragma unroll
    for (int e = 0; e < 8; ++e) a[e]     = (_Float16)gr[kb + e];
#pragma unroll
    for (int e = 0; e < 8; ++e) a[8 + e] = (_Float16)gr[kb + 16 + e];
    const int n = lane & 15, mof = (lane & 16) ? 8 : 0;
    for (int nt = wave; nt < NT; nt += 4) {
      v8f c = {};
      c = wmma16(a, load_bfrag(fih0, nt), c);
      const int col = nt * 16 + n;
      const float bias = (col < H4) ? (bih0[col] + bhh0[col]) : 0.0f;
#pragma unroll
      for (int r = 0; r < 8; ++r) sP0[(mof + r) * NCOL + col] = c[r] + bias;
    }
  }
  __syncthreads();

  const int T = *seqp;
  for (int t = 0; t < T; ++t) {
    // gates0 = p0 + h0 @ w_hh0.T
    {
      const int n = lane & 15, mof = (lane & 16) ? 8 : 0;
      for (int nt = wave; nt < NT; nt += 4) {
        v8f c = {};
#pragma unroll
        for (int kt = 0; kt < KT_H; ++kt)
          c = wmma16(make_afrag_lds(sH0, 64, kt),
                     load_bfrag(fhh0, kt * NT + nt), c);
        const int col = nt * 16 + n;
#pragma unroll
        for (int r = 0; r < 8; ++r)
          sG[(mof + r) * NCOL + col] = c[r] + sP0[(mof + r) * NCOL + col];
      }
    }
    __syncthreads();
    for (int j = tid; j < 16 * Hd; j += 128) {
      int m = j / Hd, k = j - m * Hd;
      float i_ = sigmoidf(sG[m * NCOL + k]);
      float f_ = sigmoidf(sG[m * NCOL + Hd + k]);
      float g_ = tanhf(sG[m * NCOL + 2 * Hd + k]);
      float o_ = sigmoidf(sG[m * NCOL + 3 * Hd + k]);
      float cc = f_ * sC0[m * 64 + k] + i_ * g_;
      sC0[m * 64 + k] = cc;
      sH0[m * 64 + k] = o_ * tanhf(cc);
    }
    __syncthreads();
    // gates1 = h0 @ w_ih1.T + h1 @ w_hh1.T + (b_ih1 + b_hh1)
    {
      const int n = lane & 15, mof = (lane & 16) ? 8 : 0;
      for (int nt = wave; nt < NT; nt += 4) {
        v8f c = {};
#pragma unroll
        for (int kt = 0; kt < KT_H; ++kt)
          c = wmma16(make_afrag_lds(sH0, 64, kt),
                     load_bfrag(fih1, kt * NT + nt), c);
#pragma unroll
        for (int kt = 0; kt < KT_H; ++kt)
          c = wmma16(make_afrag_lds(sH1, 64, kt),
                     load_bfrag(fhh1, kt * NT + nt), c);
        const int col = nt * 16 + n;
        const float bias = (col < H4) ? (bih1[col] + bhh1[col]) : 0.0f;
#pragma unroll
        for (int r = 0; r < 8; ++r) sG[(mof + r) * NCOL + col] = c[r] + bias;
      }
    }
    __syncthreads();
    for (int j = tid; j < 16 * Hd; j += 128) {
      int m = j / Hd, k = j - m * Hd;
      float i_ = sigmoidf(sG[m * NCOL + k]);
      float f_ = sigmoidf(sG[m * NCOL + Hd + k]);
      float g_ = tanhf(sG[m * NCOL + 2 * Hd + k]);
      float o_ = sigmoidf(sG[m * NCOL + 3 * Hd + k]);
      float cc = f_ * sC1[m * 64 + k] + i_ * g_;
      sC1[m * 64 + k] = cc;
      sH1[m * 64 + k] = o_ * tanhf(cc);
    }
    __syncthreads();
  }

  if (tid < 16 && nodeBase + tid < N) {
    float acc = fcb[0];
#pragma unroll
    for (int k = 0; k < Hd; ++k) acc += sH1[tid * 64 + k] * fcw[k];
    out[nodeBase + tid] = acc;
  }
}

// ---------------- host launcher ----------------
extern "C" void kernel_launch(void* const* d_in, const int* in_sizes, int n_in,
                              void* d_out, int out_size, void* d_ws, size_t ws_size,
                              hipStream_t stream) {
  const float* x    = (const float*)d_in[0];
  const float* Wg   = (const float*)d_in[1];
  const float* bg   = (const float*)d_in[2];
  const float* wih0 = (const float*)d_in[3];
  const float* whh0 = (const float*)d_in[4];
  const float* bih0 = (const float*)d_in[5];
  const float* bhh0 = (const float*)d_in[6];
  const float* wih1 = (const float*)d_in[7];
  const float* whh1 = (const float*)d_in[8];
  const float* bih1 = (const float*)d_in[9];
  const float* bhh1 = (const float*)d_in[10];
  const float* fcw  = (const float*)d_in[11];
  const float* fcb  = (const float*)d_in[12];
  const long long* ei = (const long long*)d_in[13];  // int64 per reference
  const int* seqp   = (const int*)d_in[14];
  float* out = (float*)d_out;

  const int N = in_sizes[0] / Fdim;
  const int E = in_sizes[13] / 2;

  char* ws = (char*)d_ws;
  size_t off = 0;
  auto take = [&](size_t bytes) -> char* {
    char* p = ws + off;
    off = (off + bytes + 255) & ~(size_t)255;
    return p;
  };
  float*     deg   = (float*)take((size_t)N * 4);                 // -> dinv in place
  float*     h     = (float*)take((size_t)N * Gdim * 4);
  float*     gnn   = (float*)take((size_t)N * Gdim * 4);
  _Float16*  fWg   = (_Float16*)take((size_t)4 * 2 * 512 * 2);
  _Float16*  fih0  = (_Float16*)take((size_t)1 * NT * 512 * 2);
  _Float16*  fhh0  = (_Float16*)take((size_t)KT_H * NT * 512 * 2);
  _Float16*  fih1  = (_Float16*)take((size_t)KT_H * NT * 512 * 2);
  _Float16*  fhh1  = (_Float16*)take((size_t)KT_H * NT * 512 * 2);

  // degrees + symmetric normalization
  init_deg<<<(N + 255) / 256, 256, 0, stream>>>(deg, N);
  count_deg<<<(E + 255) / 256, 256, 0, stream>>>(ei, deg, E);
  finalize_dinv<<<(N + 255) / 256, 256, 0, stream>>>(deg, N);

  // pack weight fragments (B layout)
  pack_b<<<(4 * 2 * 512 + 255) / 256, 256, 0, stream>>>(Wg,   fWg,  Fdim, Gdim, Gdim, 0, 4,     2);
  pack_b<<<(1 * NT * 512 + 255) / 256, 256, 0, stream>>>(wih0, fih0, Gdim, H4,   Gdim, 1, 1,     NT);
  pack_b<<<(KT_H * NT * 512 + 255) / 256, 256, 0, stream>>>(whh0, fhh0, Hd, H4, Hd, 1, KT_H, NT);
  pack_b<<<(KT_H * NT * 512 + 255) / 256, 256, 0, stream>>>(wih1, fih1, Hd, H4, Hd, 1, KT_H, NT);
  pack_b<<<(KT_H * NT * 512 + 255) / 256, 256, 0, stream>>>(whh1, fhh1, Hd, H4, Hd, 1, KT_H, NT);

  // GCN: linear -> self-loop init -> edge scatter
  gcn_mm<<<(N + 63) / 64, 128, 0, stream>>>(x, fWg, h, N);
  self_loop_bias<<<((size_t)N * Gdim + 255) / 256, 256, 0, stream>>>(h, deg, bg, gnn, N);
  edge_scatter<<<((size_t)E * 32 + 255) / 256, 256, 0, stream>>>(ei, h, deg, gnn, E);

  // fused 2-layer LSTM over T steps + FC
  lstm_mega<<<(N + 15) / 16, 128, 0, stream>>>(gnn, fih0, fhh0, fih1, fhh1,
                                               bih0, bhh0, bih1, bhh1,
                                               fcw, fcb, seqp, out, N);
}
